// OuterProductMean_18176301597121
// MI455X (gfx1250) — matrix operand order repaired and checked by reading, not compiled, not run on hardware
//
#include <hip/hip_runtime.h>

typedef _Float16 f16;
typedef _Float16 v8h  __attribute__((ext_vector_type(8)));
typedef _Float16 v16h __attribute__((ext_vector_type(16)));
typedef float    v8f  __attribute__((ext_vector_type(8)));
typedef float    v4f  __attribute__((ext_vector_type(4)));

union H16 { v16h v; v8h h[2]; };

#define N_SEQ 64
#define N_RES 512
#define C_M   64
#define C_OUT 32
#define C_Z   128

// CDNA5 async global->LDS copy (ASYNCcnt-tracked, no VGPR data round-trip).
// VGLOBAL encoding: VDST = LDS byte address, VADDR = 64-bit global address.
__device__ __forceinline__ void async_b128(const f16* g, f16* l) {
    unsigned lds = (unsigned)(size_t)l;           // low 32 bits of generic ptr = LDS offset
    unsigned long long ga = (unsigned long long)(size_t)g;
    asm volatile("global_load_async_to_lds_b128 %0, %1, off"
                 :: "v"(lds), "v"(ga) : "memory");
}
__device__ __forceinline__ void wait_async0() {
    asm volatile("s_wait_asynccnt 0" ::: "memory");
}

// ---------------------------------------------------------------------------
// Kernel 1: LayerNorm + left/right projections (+mask), store f16 transposed:
//   leftT/rightT layout [r][c][s]  (K = s contiguous -> WMMA fragment friendly)
// ---------------------------------------------------------------------------
__global__ __launch_bounds__(256)
void prep_kernel(const float* __restrict__ act, const float* __restrict__ mask,
                 const float* __restrict__ gamma, const float* __restrict__ beta,
                 const float* __restrict__ lw, const float* __restrict__ lb,
                 const float* __restrict__ rw, const float* __restrict__ rb,
                 f16* __restrict__ leftT, f16* __restrict__ rightT)
{
    __shared__ float xs[8][C_M];
    const int wave = threadIdx.x >> 5;
    const int lane = threadIdx.x & 31;
    const int row  = blockIdx.x * 8 + wave;   // flattened (s, r), 64*512 rows
    const int s = row >> 9;
    const int r = row & 511;

    const float* a = act + (size_t)row * C_M;
    float x0 = a[lane];
    float x1 = a[lane + 32];
    float sum = x0 + x1;
    #pragma unroll
    for (int off = 16; off > 0; off >>= 1) sum += __shfl_xor(sum, off, 32);
    float mu = sum * (1.0f / 64.0f);
    float d0 = x0 - mu, d1 = x1 - mu;
    float vs = d0 * d0 + d1 * d1;
    #pragma unroll
    for (int off = 16; off > 0; off >>= 1) vs += __shfl_xor(vs, off, 32);
    float rstd = rsqrtf(vs * (1.0f / 64.0f) + 1e-5f);
    xs[wave][lane]      = d0 * rstd * gamma[lane]      + beta[lane];
    xs[wave][lane + 32] = d1 * rstd * gamma[lane + 32] + beta[lane + 32];
    __syncthreads();

    const float m = mask[s * N_RES + r];
    float accL = lb[lane], accR = rb[lane];
    const float* lwr = lw + lane * C_M;
    const float* rwr = rw + lane * C_M;
    #pragma unroll 8
    for (int k = 0; k < C_M; ++k) {
        float xv = xs[wave][k];
        accL += xv * lwr[k];
        accR += xv * rwr[k];
    }
    leftT [((size_t)r * C_OUT + lane) * N_SEQ + s] = (f16)(m * accL);
    rightT[((size_t)r * C_OUT + lane) * N_SEQ + s] = (f16)(m * accR);
}

// ---------------------------------------------------------------------------
// Kernel 2: Wt[f][(c*32+e)] = (f16) output_w[c][e][f]   (A-matrix layout, GEMM2)
// ---------------------------------------------------------------------------
__global__ __launch_bounds__(256)
void wprep_kernel(const float* __restrict__ W, f16* __restrict__ Wt)
{
    int idx = blockIdx.x * 256 + threadIdx.x;   // 128*1024 = 131072 elements
    int ce = idx & 1023;
    int f  = idx >> 10;
    Wt[f * 1024 + ce] = (f16)W[ce * C_Z + f];
}

// ---------------------------------------------------------------------------
// Kernel 3: main. One block (8 waves) per 16x16 (i,j) output tile.
//   GEMM1 (transposed orientation): D rows=(jj,e) from right, cols=(i,c) from
//          left, K = s (64) -> lanes hold 8 consecutive e -> b128 pack-store.
//   GEMM2: outT[f,(i,j)] = Wt x P, K = (c,e) (1024).
// ---------------------------------------------------------------------------
__global__ __launch_bounds__(256)
void opm_kernel(const f16* __restrict__ leftT, const f16* __restrict__ rightT,
                const f16* __restrict__ Wt, const float* __restrict__ mask,
                const float* __restrict__ bias, float* __restrict__ out)
{
    __shared__ f16   Asmem[512 * 64];       // 64 KB  left  [(li*32+c)][s]
    __shared__ f16   Bsmem[2][64 * 64];     // 16 KB  right [(jj*32+e)][s], dbl-buffered
    __shared__ f16   Psmem[512 * 64];       // 64 KB  op    [(li*32+c)][(jj*32+e)]
    __shared__ float normv[16][16];         //  1 KB  reciprocal norm

    const int bi = blockIdx.x, bj = blockIdx.y;
    const int tid  = threadIdx.x;
    const int wave = tid >> 5;
    const int lane = tid & 31;
    const int hl   = lane & 15;     // row/col index inside 16
    const int lohi = lane >> 4;     // 0 = lanes 0-15, 1 = lanes 16-31

    // ---- kick off async staging: A tile (64 KB) + right slab 0 (8 KB) ----
    {
        const f16* asrc = leftT + (size_t)bi * 16 * C_OUT * N_SEQ;
        #pragma unroll
        for (int t = 0; t < 16; ++t)
            async_b128(asrc + (tid + t * 256) * 8, &Asmem[(tid + t * 256) * 8]);
        const f16* bsrc = rightT + (size_t)(bj * 16) * C_OUT * N_SEQ;
        #pragma unroll
        for (int t = 0; t < 2; ++t)
            async_b128(bsrc + (tid + t * 256) * 8, &Bsmem[0][(tid + t * 256) * 8]);
    }
    // L2-resident weight rows this wave will stream 8x
    __builtin_prefetch(Wt + (size_t)(wave * 16 + hl) * 1024, 0, 2);

    // ---- reciprocal norm for all 256 (li, lj) pairs (overlaps async) ----
    {
        int li = tid >> 4, lj = tid & 15;
        int gi = bi * 16 + li, gj = bj * 16 + lj;
        float acc = 0.f;
        #pragma unroll 8
        for (int s = 0; s < N_SEQ; ++s)
            acc += mask[s * N_RES + gi] * mask[s * N_RES + gj];
        normv[li][lj] = 1.0f / (1e-3f + acc);
    }

    for (int g = 0; g < 8; ++g) {       // 8 slabs of 2 j-columns each
        wait_async0();                  // this wave's async copies landed
        __syncthreads();                // everyone's landed; Psmem consumers done
        const f16* Bs = Bsmem[g & 1];

        // ================= GEMM1: P^T tiles (K = 64) =================
        // B operand (cols = (i,c)) fragments: loaded once per slab per wave
        H16 bfrag[4][2];
        #pragma unroll
        for (int cq = 0; cq < 4; ++cq) {
            const int ct = wave * 4 + cq;
            #pragma unroll
            for (int kk = 0; kk < 2; ++kk) {   // B frag: lane n = col, K chunks 0..7 / 16..23 (+8 hi)
                const f16* p = &Asmem[(ct * 16 + hl) * 64 + kk * 32 + lohi * 8];
                bfrag[cq][kk].h[0] = *(const v8h*)(p);
                bfrag[cq][kk].h[1] = *(const v8h*)(p + 16);
            }
        }
        #pragma unroll
        for (int rt = 0; rt < 4; ++rt) {       // rows = (jj,e), 4 row-tiles
            H16 afrag[2];
            #pragma unroll
            for (int kk = 0; kk < 2; ++kk) {   // A frag: K chunks 0..7 / 16..23 (+8 hi)
                const f16* p = &Bs[(rt * 16 + hl) * 64 + kk * 32 + lohi * 8];
                afrag[kk].h[0] = *(const v8h*)(p);
                afrag[kk].h[1] = *(const v8h*)(p + 16);
            }
            #pragma unroll
            for (int cq = 0; cq < 4; ++cq) {
                const int ct = wave * 4 + cq;
                v8f acc = {};
                acc = __builtin_amdgcn_wmma_f32_16x16x32_f16(false, afrag[0].v, false,
                        bfrag[cq][0].v, (short)0, acc, false, false);
                acc = __builtin_amdgcn_wmma_f32_16x16x32_f16(false, afrag[1].v, false,
                        bfrag[cq][1].v, (short)0, acc, false, false);
                // lane: col (i,c) fixed, 8 consecutive e -> contiguous b128 store
                v8h ph;
                #pragma unroll
                for (int v = 0; v < 8; ++v) ph[v] = (f16)acc[v];
                *(v8h*)&Psmem[(ct * 16 + hl) * 64 + rt * 16 + lohi * 8] = ph;
            }
        }
        __syncthreads();                // Psmem ready; Bs reads done

        // overlap: fetch next right slab while GEMM2 runs
        if (g < 7) {
            const f16* bsrc = rightT + (size_t)(bj * 16 + (g + 1) * 2) * C_OUT * N_SEQ;
            #pragma unroll
            for (int t = 0; t < 2; ++t)
                async_b128(bsrc + (tid + t * 256) * 8,
                           &Bsmem[(g + 1) & 1][(tid + t * 256) * 8]);
        }

        // ============ GEMM2: outT[f,(i,j)] = Wt x P (K = 1024) ============
        v8f acc2[2] = {{}, {}};
        const f16* wrow = Wt + (size_t)(wave * 16 + hl) * 1024;
        #pragma unroll 4
        for (int k = 0; k < 32; ++k) {
            const int k0 = k * 32;              // K chunk: c = k, e = 0..31
            H16 af2;
            af2.h[0] = *(const v8h*)(wrow + k0 + lohi * 8);
            af2.h[1] = *(const v8h*)(wrow + k0 + lohi * 8 + 16);
            #pragma unroll
            for (int ct2 = 0; ct2 < 2; ++ct2) {
                const int p  = ct2 * 16 + hl;
                const int li = p >> 1, jj = p & 1;
                const f16* brow = &Psmem[(li * 32 + k) * 64 + jj * 32 + lohi * 8];
                H16 bf2;
                bf2.h[0] = *(const v8h*)(brow);
                bf2.h[1] = *(const v8h*)(brow + 16);
                acc2[ct2] = __builtin_amdgcn_wmma_f32_16x16x32_f16(false, af2.v, false,
                              bf2.v, (short)0, acc2[ct2], false, false);
            }
        }
        // ---- epilogue: bias + reciprocal-norm scale, 2x b128 stores ----
        #pragma unroll
        for (int ct2 = 0; ct2 < 2; ++ct2) {
            const int p  = ct2 * 16 + hl;
            const int li = p >> 1, jj = p & 1;
            const int lj = g * 2 + jj;
            const float scale = normv[li][lj];
            const int i = bi * 16 + li, j = bj * 16 + lj;
            const int fbase = wave * 16 + lohi * 8;
            float* o = out + ((size_t)i * N_RES + j) * C_Z + fbase;
            v4f r0, r1;
            #pragma unroll
            for (int v = 0; v < 4; ++v) r0[v] = (acc2[ct2][v]     + bias[fbase + v])     * scale;
            #pragma unroll
            for (int v = 0; v < 4; ++v) r1[v] = (acc2[ct2][v + 4] + bias[fbase + 4 + v]) * scale;
            *(v4f*)o       = r0;
            *(v4f*)(o + 4) = r1;
        }
    }
}

// ---------------------------------------------------------------------------
extern "C" void kernel_launch(void* const* d_in, const int* in_sizes, int n_in,
                              void* d_out, int out_size, void* d_ws, size_t ws_size,
                              hipStream_t stream) {
    const float* act   = (const float*)d_in[0];
    const float* mask  = (const float*)d_in[1];
    const float* gamma = (const float*)d_in[2];
    const float* beta  = (const float*)d_in[3];
    const float* lw    = (const float*)d_in[4];
    const float* lb    = (const float*)d_in[5];
    const float* rw    = (const float*)d_in[6];
    const float* rb    = (const float*)d_in[7];
    const float* ow    = (const float*)d_in[8];
    const float* ob    = (const float*)d_in[9];
    float* out = (float*)d_out;

    f16* leftT  = (f16*)d_ws;                       // 512*32*64 halves (2 MB)
    f16* rightT = leftT  + (size_t)N_RES * C_OUT * N_SEQ;
    f16* Wt     = rightT + (size_t)N_RES * C_OUT * N_SEQ;  // 1024*128 halves

    prep_kernel<<<(N_SEQ * N_RES) / 8, 256, 0, stream>>>(act, mask, gamma, beta,
                                                         lw, lb, rw, rb, leftT, rightT);
    wprep_kernel<<<(C_OUT * C_OUT * C_Z) / 256, 256, 0, stream>>>(ow, Wt);
    dim3 grid(N_RES / 16, N_RES / 16);
    opm_kernel<<<grid, 256, 0, stream>>>(leftT, rightT, Wt, mask, ob, out);
}
